// CGC_72481868087299
// MI455X (gfx1250) — matrix operand order repaired and checked by reading, not compiled, not run on hardware
//
#include <hip/hip_runtime.h>

typedef __attribute__((ext_vector_type(16))) __bf16 v16bf;
typedef __attribute__((ext_vector_type(8)))  float  v8f;

// Sizes
// B=65536, D=256, H=128, O=64, 16 experts (12 task-specific + 4 shared), NT=3, NE=8

__device__ __forceinline__ unsigned short f2bf(float f) {
  unsigned u = __float_as_uint(f);
  return (unsigned short)((u + 0x7FFFu + ((u >> 16) & 1u)) >> 16);  // RNE
}
__device__ __forceinline__ float bf2f(unsigned short b) {
  return __uint_as_float(((unsigned)b) << 16);
}

#define WMMA_BF16(a, b, c) \
  __builtin_amdgcn_wmma_f32_16x16x32_bf16(false, (a), false, (b), (short)0, (c), false, false)

// ---------------------------------------------------------------------------
// Weight prep: fp32 -> bf16 hi/lo planes, pre-swizzled WMMA B-frag layout.
// B-frag (32x16 bf16): lane l holds column c = l&15; koff = (l<16)?0:8;
// j-th bf16 of the lane = W[kt*32 + koff + (j<8?j:j+8)][c].
// Storage: frag*1024 + plane*512 + lane*16 + j  (ushort elements)
// ---------------------------------------------------------------------------
__global__ __launch_bounds__(256) void cgc_prep_w(
    const float* __restrict__ w1t, const float* __restrict__ w1s,
    const float* __restrict__ w2t, const float* __restrict__ w2s,
    unsigned short* __restrict__ w1f, unsigned short* __restrict__ w2f)
{
  int idx = blockIdx.x * 256 + threadIdx.x;
  const int N1 = 16 * 8 * 8 * 512;  // 16 experts, 8 ktiles, 8 ntiles, 512 lane-elems
  if (idx < N1) {
    int j = idx & 15, lane = (idx >> 4) & 31, nt = (idx >> 9) & 7, kt = (idx >> 12) & 7, e = idx >> 15;
    int k = kt * 32 + ((lane >> 4) << 3) + ((j < 8) ? j : j + 8);
    int n = nt * 16 + (lane & 15);
    float v = (e < 12) ? w1t[(e * 256 + k) * 128 + n] : w1s[((e - 12) * 256 + k) * 128 + n];
    unsigned short hi = f2bf(v);
    unsigned short lo = f2bf(v - bf2f(hi));
    int f = (e * 8 + kt) * 8 + nt;
    w1f[f * 1024 + (idx & 511)] = hi;
    w1f[f * 1024 + 512 + (idx & 511)] = lo;
  } else {
    int id2 = idx - N1;
    if (id2 >= 16 * 4 * 4 * 512) return;
    int j = id2 & 15, lane = (id2 >> 4) & 31, nt = (id2 >> 9) & 3, kt = (id2 >> 11) & 3, e = id2 >> 13;
    int k = kt * 32 + ((lane >> 4) << 3) + ((j < 8) ? j : j + 8);
    int n = nt * 16 + (lane & 15);
    float v = (e < 12) ? w2t[(e * 128 + k) * 64 + n] : w2s[((e - 12) * 128 + k) * 64 + n];
    unsigned short hi = f2bf(v);
    unsigned short lo = f2bf(v - bf2f(hi));
    int f = (e * 4 + kt) * 4 + nt;
    w2f[f * 1024 + (id2 & 511)] = hi;
    w2f[f * 1024 + 512 + (id2 & 511)] = lo;
  }
}

// ---------------------------------------------------------------------------
// x prep: fp32 -> bf16 hi/lo planes, WMMA A-frag layout (16-bit A 16x32):
// lane l: row M = l&15; koff = (l<16)?0:8; j<8 -> K = koff+j ; j>=8 -> K = koff+j+8
// ---------------------------------------------------------------------------
__global__ __launch_bounds__(256) void cgc_prep_x(
    const float* __restrict__ x, unsigned short* __restrict__ xfrag)
{
  int idx = blockIdx.x * 256 + threadIdx.x;  // < 4096*8*512
  int j = idx & 15, lane = (idx >> 4) & 31, kt = (idx >> 9) & 7, rb = idx >> 12;
  int row = rb * 16 + (lane & 15);
  int k = kt * 32 + ((lane >> 4) << 3) + ((j < 8) ? j : j + 8);
  float v = x[row * 256 + k];
  unsigned short hi = f2bf(v);
  unsigned short lo = f2bf(v - bf2f(hi));
  int f = rb * 8 + kt;
  xfrag[f * 1024 + (idx & 511)] = hi;
  xfrag[f * 1024 + 512 + (idx & 511)] = lo;
}

// ---------------------------------------------------------------------------
// Gating: logits -> softmax over 8 experts, for 3 tasks.  [3][B][8] fp32
// ---------------------------------------------------------------------------
__global__ __launch_bounds__(256) void cgc_gates(
    const float* __restrict__ x, const float* __restrict__ gw,
    const float* __restrict__ gb, float* __restrict__ gates)
{
  int row = blockIdx.x * 256 + threadIdx.x;
  float acc[24];
#pragma unroll
  for (int i = 0; i < 24; ++i) acc[i] = gb[i];
  const float* xr = x + row * 256;
  for (int d = 0; d < 256; ++d) {
    float xv = xr[d];
#pragma unroll
    for (int t = 0; t < 3; ++t)
#pragma unroll
      for (int e = 0; e < 8; ++e)
        acc[t * 8 + e] = fmaf(xv, gw[(t * 256 + d) * 8 + e], acc[t * 8 + e]);
  }
#pragma unroll
  for (int t = 0; t < 3; ++t) {
    float m = acc[t * 8];
#pragma unroll
    for (int e = 1; e < 8; ++e) m = fmaxf(m, acc[t * 8 + e]);
    float ex[8];
    float s = 0.f;
#pragma unroll
    for (int e = 0; e < 8; ++e) { ex[e] = __expf(acc[t * 8 + e] - m); s += ex[e]; }
    float inv = 1.0f / s;
#pragma unroll
    for (int e = 0; e < 8; ++e) gates[(t * 65536 + row) * 8 + e] = ex[e] * inv;
  }
}

// ---------------------------------------------------------------------------
// One expert FFN for a 16-row block: GEMM1 (split bf16) -> relu -> h stored
// to LDS *directly in GEMM2 A-fragment order* -> GEMM2 (split bf16).
// Returns acc2[4] (16x64 tile, C-layout), bias b2 NOT yet applied.
//
// A-frag LDS layout per plane (2048 ushort per wave):
//   offset = kt*512 + lane*16 + j  holds  h[lane&15][kt*32 + ((lane>>4)<<3) + (j<8?j:j+8)]
// Producer mapping for C-layout element (row = half*8+r, col = nt*16+ln):
//   kt = nt>>1 ; j = (ln&7) + (nt&1)*8 ; lane_t = row + ((ln&8)<<1)
// ---------------------------------------------------------------------------
__device__ __forceinline__ void cgc_expert(
    int e, int rb, int lane, int half, int ln,
    const unsigned short* __restrict__ xfrag,
    const unsigned short* __restrict__ w1f,
    const unsigned short* __restrict__ w2f,
    const float* __restrict__ b1p,
    unsigned short* __restrict__ Hh, unsigned short* __restrict__ Hl,
    v8f acc2[4])
{
  const v8f zacc = {0.f, 0.f, 0.f, 0.f, 0.f, 0.f, 0.f, 0.f};
  v8f acc1[8];
#pragma unroll
  for (int n = 0; n < 8; ++n) acc1[n] = zacc;

  // GEMM1: (16x256) x (256x128), split hi/lo: 8kt x 8nt x 3 WMMA
#pragma unroll
  for (int kt = 0; kt < 8; ++kt) {
    const unsigned short* xp = xfrag + (rb * 8 + kt) * 1024 + lane * 16;
    v16bf ah = *(const v16bf*)xp;
    v16bf al = *(const v16bf*)(xp + 512);
#pragma unroll
    for (int nt = 0; nt < 8; ++nt) {
      const unsigned short* wp = w1f + ((e * 8 + kt) * 8 + nt) * 1024 + lane * 16;
      v16bf bh = *(const v16bf*)wp;
      v16bf bl = *(const v16bf*)(wp + 512);
      acc1[nt] = WMMA_BF16(ah, bh, acc1[nt]);
      acc1[nt] = WMMA_BF16(ah, bl, acc1[nt]);
      acc1[nt] = WMMA_BF16(al, bh, acc1[nt]);
    }
  }

  // bias + relu + split-store h directly in GEMM2 A-frag order
  const int lhi = (ln & 8) << 1;   // 0 or 16: destination lane group
  const int lj  = ln & 7;
#pragma unroll
  for (int nt = 0; nt < 8; ++nt) {
    float bv = b1p[nt * 16 + ln];
    const int j = lj + ((nt & 1) << 3);
    const int ktbase = (nt >> 1) * 512;
#pragma unroll
    for (int r = 0; r < 8; ++r) {
      float hv = fmaxf(acc1[nt][r] + bv, 0.f);
      unsigned short hb = f2bf(hv);
      unsigned short lb = f2bf(hv - bf2f(hb));
      int off = ktbase + (lhi + half * 8 + r) * 16 + j;
      Hh[off] = hb;
      Hl[off] = lb;
    }
  }

  // GEMM2: (16x128) x (128x64), split hi/lo: 4kt x 4nt x 3 WMMA
#pragma unroll
  for (int n = 0; n < 4; ++n) acc2[n] = zacc;
#pragma unroll
  for (int kt = 0; kt < 4; ++kt) {
    v16bf ah = *(const v16bf*)(Hh + kt * 512 + lane * 16);
    v16bf al = *(const v16bf*)(Hl + kt * 512 + lane * 16);
#pragma unroll
    for (int nt = 0; nt < 4; ++nt) {
      const unsigned short* wp = w2f + ((e * 4 + kt) * 4 + nt) * 1024 + lane * 16;
      v16bf bh = *(const v16bf*)wp;
      v16bf bl = *(const v16bf*)(wp + 512);
      acc2[nt] = WMMA_BF16(ah, bh, acc2[nt]);
      acc2[nt] = WMMA_BF16(ah, bl, acc2[nt]);
      acc2[nt] = WMMA_BF16(al, bh, acc2[nt]);
    }
  }
}

// ---------------------------------------------------------------------------
// Main fused kernel: 1 wave (wave32) per 16-row block, 4 waves per block.
// ---------------------------------------------------------------------------
__global__ __launch_bounds__(128) void cgc_main(
    const unsigned short* __restrict__ xfrag,
    const unsigned short* __restrict__ w1f,
    const unsigned short* __restrict__ w2f,
    const float* __restrict__ gates,
    const float* __restrict__ b1t, const float* __restrict__ b1s,
    const float* __restrict__ b2t, const float* __restrict__ b2s,
    float* __restrict__ out)
{
  __shared__ unsigned short Hhi[4][16 * 128];
  __shared__ unsigned short Hlo[4][16 * 128];
  const int wid = threadIdx.x >> 5, lane = threadIdx.x & 31;
  const int half = lane >> 4, ln = lane & 15;
  const int rb = blockIdx.x * 4 + wid;
  unsigned short* Hh = Hhi[wid];
  unsigned short* Hl = Hlo[wid];
  const int rowbase = rb * 16 + half * 8;

  float oacc[3][4][8];
#pragma unroll
  for (int t = 0; t < 3; ++t)
#pragma unroll
    for (int n = 0; n < 4; ++n)
#pragma unroll
      for (int r = 0; r < 8; ++r) oacc[t][n][r] = 0.f;

  v8f acc2[4];

  // Task-specific experts: expert e = t*4+s feeds only task t (gate slot s).
#pragma unroll
  for (int t = 0; t < 3; ++t) {
    for (int s = 0; s < 4; ++s) {
      const int e = t * 4 + s;
      cgc_expert(e, rb, lane, half, ln, xfrag, w1f, w2f, b1t + e * 128, Hh, Hl, acc2);
      const float* b2p = b2t + e * 64;
      float g[8];
#pragma unroll
      for (int r = 0; r < 8; ++r) g[r] = gates[(t * 65536 + rowbase + r) * 8 + s];
#pragma unroll
      for (int nt = 0; nt < 4; ++nt) {
        float bv = b2p[nt * 16 + ln];
#pragma unroll
        for (int r = 0; r < 8; ++r) {
          float ov = acc2[nt][r] + bv;
          oacc[t][nt][r] = fmaf(g[r], ov, oacc[t][nt][r]);
        }
      }
    }
  }

  // Shared experts: e = 12..15 feed all 3 tasks (gate slot 4 + (e-12)).
  for (int es = 0; es < 4; ++es) {
    const int e = 12 + es;
    cgc_expert(e, rb, lane, half, ln, xfrag, w1f, w2f, b1s + es * 128, Hh, Hl, acc2);
    const float* b2p = b2s + es * 64;
    const int ge = 4 + es;
    float g0[8], g1[8], g2[8];
#pragma unroll
    for (int r = 0; r < 8; ++r) {
      g0[r] = gates[(0 * 65536 + rowbase + r) * 8 + ge];
      g1[r] = gates[(1 * 65536 + rowbase + r) * 8 + ge];
      g2[r] = gates[(2 * 65536 + rowbase + r) * 8 + ge];
    }
#pragma unroll
    for (int nt = 0; nt < 4; ++nt) {
      float bv = b2p[nt * 16 + ln];
#pragma unroll
      for (int r = 0; r < 8; ++r) {
        float ov = acc2[nt][r] + bv;
        oacc[0][nt][r] = fmaf(g0[r], ov, oacc[0][nt][r]);
        oacc[1][nt][r] = fmaf(g1[r], ov, oacc[1][nt][r]);
        oacc[2][nt][r] = fmaf(g2[r], ov, oacc[2][nt][r]);
      }
    }
  }

  // Write 3 task outputs ([B,64] each, concatenated).
#pragma unroll
  for (int t = 0; t < 3; ++t)
#pragma unroll
    for (int nt = 0; nt < 4; ++nt)
#pragma unroll
      for (int r = 0; r < 8; ++r)
        out[(t * 65536 + rowbase + r) * 64 + nt * 16 + ln] = oacc[t][nt][r];
}

// ---------------------------------------------------------------------------
extern "C" void kernel_launch(void* const* d_in, const int* in_sizes, int n_in,
                              void* d_out, int out_size, void* d_ws, size_t ws_size,
                              hipStream_t stream) {
  const float* x   = (const float*)d_in[0];
  const float* w1t = (const float*)d_in[1];
  const float* b1t = (const float*)d_in[2];
  const float* w2t = (const float*)d_in[3];
  const float* b2t = (const float*)d_in[4];
  const float* w1s = (const float*)d_in[5];
  const float* b1s = (const float*)d_in[6];
  const float* w2s = (const float*)d_in[7];
  const float* b2s = (const float*)d_in[8];
  const float* gw  = (const float*)d_in[9];
  const float* gb  = (const float*)d_in[10];
  float* out = (float*)d_out;
  (void)in_sizes; (void)n_in; (void)out_size; (void)ws_size;

  // Workspace layout (bytes):
  //   xfrag : 4096 rowblocks * 8 ktiles * 1024 ushort  = 67,108,864
  //   w1f   : 16 e * 64 frags * 1024 ushort            =  2,097,152
  //   w2f   : 16 e * 16 frags * 1024 ushort            =    524,288
  //   gates : 3 * 65536 * 8 fp32                       =  6,291,456
  char* ws = (char*)d_ws;
  unsigned short* xfrag = (unsigned short*)(ws);
  unsigned short* w1f   = (unsigned short*)(ws + (size_t)67108864);
  unsigned short* w2f   = (unsigned short*)(ws + (size_t)67108864 + 2097152);
  float* gates          = (float*)(ws + (size_t)67108864 + 2097152 + 524288);

  cgc_prep_w<<<2560, 256, 0, stream>>>(w1t, w1s, w2t, w2s, w1f, w2f);
  cgc_prep_x<<<65536, 256, 0, stream>>>(x, xfrag);
  cgc_gates<<<256, 256, 0, stream>>>(x, gw, gb, gates);
  cgc_main<<<1024, 128, 0, stream>>>(xfrag, w1f, w2f, gates, b1t, b1s, b2t, b2s, out);
}
